// MultiheadSelfAttention_32993938768107
// MI455X (gfx1250) — compile-verified, matrix operand order checked by reading, not run on hardware
//
#include <hip/hip_runtime.h>

// ---------- problem constants ----------
#define DDIM 1024
#define NSEQ 8192
#define BSZ  2
#define HEADS 16
#define DKH  64
#define CH   256   // chunk size S
#define NCH  32    // chunks M

typedef __attribute__((ext_vector_type(16))) __bf16 v16bf;
typedef __attribute__((ext_vector_type(8)))  __bf16 v8bf;
typedef __attribute__((ext_vector_type(8)))  float  v8f;
typedef __attribute__((ext_vector_type(8)))  unsigned short u16x8;
typedef __attribute__((ext_vector_type(4)))  unsigned int  u32x4;
typedef __attribute__((ext_vector_type(8)))  unsigned int  u32x8;

__device__ __forceinline__ v8f vzero() {
  v8f v;
#pragma unroll
  for (int i = 0; i < 8; ++i) v[i] = 0.0f;
  return v;
}

// f32 -> bf16 (round to nearest even)
__device__ __forceinline__ unsigned short f2bf(float f) {
  unsigned u = __float_as_uint(f);
  unsigned r = u + 0x7FFFu + ((u >> 16) & 1u);
  return (unsigned short)(r >> 16);
}

// ---- 16-lane max butterfly in pure VALU via DPP16 (no LDS traffic) ----
__device__ __forceinline__ float red_max16(float v) {
  v = fmaxf(v, __int_as_float(__builtin_amdgcn_mov_dpp(__float_as_int(v), 0xB1, 0xF, 0xF, true)));
  v = fmaxf(v, __int_as_float(__builtin_amdgcn_mov_dpp(__float_as_int(v), 0x4E, 0xF, 0xF, true)));
  v = fmaxf(v, __int_as_float(__builtin_amdgcn_mov_dpp(__float_as_int(v), 0x141, 0xF, 0xF, true)));
  v = fmaxf(v, __int_as_float(__builtin_amdgcn_mov_dpp(__float_as_int(v), 0x140, 0xF, 0xF, true)));
  return v;
}

// Load a 16-bit A/B WMMA fragment. Per ISA layout the lane's data is two 16-byte
// blocks: lo = K[kh*8 .. +7], hi = K[16+kh*8 .. +7]; `p` = rowstart + kh*8.
__device__ __forceinline__ v16bf ld_frag(const unsigned short* p) {
  v16bf r;
  *(v8bf*)&r = *(const v8bf*)p;
  *((v8bf*)&r + 1) = *(const v8bf*)(p + 16);
  return r;
}

__device__ __forceinline__ v8f wmma_bf16(v16bf a, v16bf b, v8f c) {
  return __builtin_amdgcn_wmma_f32_16x16x32_bf16(false, a, false, b,
                                                 (short)0, c, false, false);
}

// constant all-ones 32x16 B fragment: rowsum(P) = P @ ones via one WMMA
__device__ __forceinline__ v16bf ones_frag() {
  v16bf r;
#pragma unroll
  for (int i = 0; i < 16; ++i) r[i] = (__bf16)1.0f;
  return r;
}

// ---------- kernel 0: convert x to bf16 (streaming, vectorized) ----------
__global__ __launch_bounds__(256) void conv_x(const float* __restrict__ x,
                                              unsigned short* __restrict__ xh) {
  const size_t i0 = ((size_t)blockIdx.x * 256 + threadIdx.x) * 8;
  u16x8 o;
#pragma unroll
  for (int i = 0; i < 8; ++i) o[i] = f2bf(x[i0 + i]);
  *(u16x8*)(xh + i0) = o;
}

// ---------- kernel 1: transpose + convert weights to bf16, n-major ----------
__global__ __launch_bounds__(256) void prep_weights(
    const float* __restrict__ Wq, const float* __restrict__ Wk,
    const float* __restrict__ Wv, const float* __restrict__ Wo,
    unsigned short* __restrict__ Wt) {
  const int g = blockIdx.y;
  const float* W = (g == 0) ? Wq : ((g == 1) ? Wk : ((g == 2) ? Wv : Wo));
  const int flat = blockIdx.x * 256 + threadIdx.x;
  const int n = flat >> 10, k = flat & 1023;
  Wt[(size_t)g * DDIM * DDIM + (size_t)n * DDIM + k] = f2bf(W[(size_t)k * DDIM + n]);
}

// ---------- GEMM: C[16384 x 1024] = A_bf16[16384 x 1024] @ W, + bias ----------
// MODE 0: out = bf16 scattered into [g][B,H,M,S,DK] (g = blockIdx.z)
// MODE 1: out = f32 row-major d_out, g fixed = 3 (Wo)
// B tile (weights) is staged into LDS by the Tensor Data Mover: one
// tensor_load_to_lds per k-step, 32x128 bf16 tile, TDM pad feature recreates
// the TSTR=40 padded row layout (16 DWORDs data + 4 DWORDs pad per row).
#define TSTR 40  // LDS row stride in elements (80 B, multiple of 16 B)

template <int MODE>
__global__ __launch_bounds__(256) void gemm_kernel(
    const unsigned short* __restrict__ Abf, const unsigned short* __restrict__ Wt,
    const float* __restrict__ bias0, const float* __restrict__ bias1,
    const float* __restrict__ bias2, unsigned short* __restrict__ out_bf,
    float* __restrict__ out_f) {
  __shared__ unsigned short Asm[128 * TSTR];
  __shared__ unsigned short Bsm[128 * TSTR];

  const int tid  = threadIdx.x;
  const int lane = tid & 31;
  const int wave = tid >> 5;
  const int wm = wave & 3;   // 4 row-blocks of 32 rows
  const int wn = wave >> 2;  // 2 col-blocks of 64 cols
  const int n0 = blockIdx.x * 128;
  const int m0 = blockIdx.y * 128;
  const int g = (MODE == 0) ? blockIdx.z : 3;
  const float* bias =
      (MODE == 0) ? ((g == 0) ? bias0 : ((g == 1) ? bias1 : bias2)) : bias0;
  const unsigned short* Wg = Wt + (size_t)g * DDIM * DDIM;

  v8f acc[2][4];
#pragma unroll
  for (int i = 0; i < 2; ++i)
#pragma unroll
    for (int j = 0; j < 4; ++j) acc[i][j] = vzero();

  const int srow = tid >> 1;          // 0..127
  const int sc0  = (tid & 1) * 16;    // 0 or 16
  const unsigned bsm_lds = (unsigned)(unsigned long long)(uintptr_t)&Bsm[0];

  for (int kk = 0; kk < DDIM; kk += 32) {
    __syncthreads();
    // ---- stage A tile (128 x 32 bf16) via vector copies ----
    {
      const unsigned short* ap = Abf + (size_t)(m0 + srow) * DDIM + kk + sc0;
      uint4 d0 = *(const uint4*)ap;
      uint4 d1 = *(const uint4*)(ap + 8);
      unsigned short* dst = Asm + srow * TSTR + sc0;
      *(uint4*)dst = d0;
      *(uint4*)(dst + 8) = d1;
    }
    // ---- stage B tile (128 cols x 32 k) via Tensor Data Mover ----
    if (wave == 0) {
      const unsigned long long ga =
          (unsigned long long)(uintptr_t)(Wg + (size_t)n0 * DDIM + kk);
      u32x4 g0;
      g0[0] = 1u;                                  // count=1, user descriptor
      g0[1] = bsm_lds;                             // lds_addr (bytes)
      g0[2] = (unsigned)ga;                        // global_addr[31:0]
      g0[3] = (unsigned)(ga >> 32) | 0x80000000u;  // addr[56:32] | type=2
      u32x8 g1;
      // data_size=2B (1<<16) | pad_enable (1<<20) | pad_interval=16DW (3<<22)
      // | pad_amount=4DW (3<<25); workgroup_mask=0
      g1[0] = (1u << 16) | (1u << 20) | (3u << 22) | (3u << 25);
      g1[1] = (1024u & 0xFFFFu) << 16;   // tensor_dim0[15:0] in [31:16]
      g1[2] = (0u) | (1024u << 16);      // tensor_dim0[31:16] | tensor_dim1[15:0]
      g1[3] = (0u) | (32u << 16);        // tensor_dim1[31:16] | tile_dim0=32
      g1[4] = 128u;                      // tile_dim1=128 | tile_dim2=0
      g1[5] = 1024u;                     // tensor_dim0_stride[31:0]
      g1[6] = 0u;                        // dim0_stride[47:32] | dim1_stride[15:0]
      g1[7] = 0u;                        // dim1_stride[47:16]
      asm volatile("tensor_load_to_lds %0, %1" : : "s"(g0), "s"(g1) : "memory");
      __builtin_amdgcn_s_wait_tensorcnt(0);
    }
    __syncthreads();

    v16bf af[2];
#pragma unroll
    for (int rt = 0; rt < 2; ++rt)
      af[rt] = ld_frag(Asm + (wm * 32 + rt * 16 + (lane & 15)) * TSTR + (lane >> 4) * 8);
#pragma unroll
    for (int ct = 0; ct < 4; ++ct) {
      v16bf bf = ld_frag(Bsm + (wn * 64 + ct * 16 + (lane & 15)) * TSTR + (lane >> 4) * 8);
#pragma unroll
      for (int rt = 0; rt < 2; ++rt) acc[rt][ct] = wmma_bf16(af[rt], bf, acc[rt][ct]);
    }
  }

  // ---- epilogue ----
#pragma unroll
  for (int rt = 0; rt < 2; ++rt)
#pragma unroll
    for (int ct = 0; ct < 4; ++ct) {
      const int col = n0 + wn * 64 + ct * 16 + (lane & 15);
      const float bv = bias[col];
#pragma unroll
      for (int r = 0; r < 8; ++r) {
        const int row = m0 + wm * 32 + rt * 16 + r + 8 * (lane >> 4);
        const float val = acc[rt][ct][r] + bv;
        if (MODE == 0) {
          const int b = row >> 13, n = row & (NSEQ - 1);
          const int h = col >> 6, dk = col & 63;
          const int m = n >> 8, s = n & 255;
          const size_t chunk = (size_t)((b * HEADS + h) * NCH + m);
          out_bf[(size_t)g * ((size_t)BSZ * HEADS * NCH * CH * DKH) +
                 chunk * CH * DKH + (size_t)s * DKH + dk] = f2bf(val);
        } else {
          out_f[(size_t)row * DDIM + col] = val;
        }
      }
    }
}

// ---------- kernel 3: flash-style block-local attention, 1 WG per (b,h,m) ----------
#define VSTR 264  // Vt row stride (528 B, multiple of 16 B)

__global__ __launch_bounds__(256) void attn_kernel(
    const unsigned short* __restrict__ Q, const unsigned short* __restrict__ K,
    const unsigned short* __restrict__ V, unsigned short* __restrict__ out_bf) {
  __shared__ unsigned short Vt[64 * VSTR];   // V transposed: [dk][t]
  __shared__ unsigned short pbuf[8 * 512];   // per-wave P tile relayout buffer

  const int tid = threadIdx.x;
  const int lane = tid & 31;
  const int wave = tid >> 5;
  const int cidx = blockIdx.x;                 // (b*16 + h)*32 + m
  const int m = cidx & (NCH - 1);
  const int h = (cidx >> 5) & (HEADS - 1);
  const int b = cidx >> 9;
  const size_t base = (size_t)cidx * CH * DKH;

  // stage V transposed into LDS
#pragma unroll
  for (int i = 0; i < 8; ++i) {
    const int flat0 = i * 2048 + tid * 8;      // element index, 16-B aligned
    const int s = flat0 >> 6;
    const int dk0 = flat0 & 63;
    u16x8 vv = *(const u16x8*)(V + base + flat0);
#pragma unroll
    for (int j = 0; j < 8; ++j) Vt[(dk0 + j) * VSTR + s] = vv[j];
  }
  __syncthreads();

  unsigned short* pw = pbuf + wave * 512;
  const v16bf onesf = ones_frag();

  for (int rt = 0; rt < 2; ++rt) {
    const int rowbase = rt * 128 + wave * 16;  // this wave's 16 query rows
    v16bf qf[2];
#pragma unroll
    for (int kf = 0; kf < 2; ++kf)
      qf[kf] = ld_frag(Q + base + (size_t)(rowbase + (lane & 15)) * DKH +
                       kf * 32 + (lane >> 4) * 8);

    float m_run[8], l_run[8];
#pragma unroll
    for (int r = 0; r < 8; ++r) { m_run[r] = -3.0e38f; l_run[r] = 0.0f; }
    v8f o[4];
#pragma unroll
    for (int c = 0; c < 4; ++c) o[c] = vzero();

    for (int kt = 0; kt < 8; ++kt) {
      const int t0 = kt * 32;
      v8f s0 = vzero(), s1 = vzero();
#pragma unroll
      for (int kf = 0; kf < 2; ++kf) {
        v16bf k0 = ld_frag(K + base + (size_t)(t0 + (lane & 15)) * DKH +
                           kf * 32 + (lane >> 4) * 8);
        v16bf k1 = ld_frag(K + base + (size_t)(t0 + 16 + (lane & 15)) * DKH +
                           kf * 32 + (lane >> 4) * 8);
        s0 = wmma_bf16(qf[kf], k0, s0);
        s1 = wmma_bf16(qf[kf], k1, s1);
      }
      // online softmax (row = per-VGPR across each 16-lane half).
      // Row max via DPP butterfly; row sum comes later from P @ ones (WMMA).
      float p0[8], p1[8];
#pragma unroll
      for (int r = 0; r < 8; ++r) {
        const float a0 = s0[r] * 0.125f;  // 1/sqrt(64)
        const float a1 = s1[r] * 0.125f;
        const float mx = red_max16(fmaxf(a0, a1));
        const float mn = fmaxf(m_run[r], mx);
        const float alpha = __expf(m_run[r] - mn);
        p0[r] = __expf(a0 - mn);
        p1[r] = __expf(a1 - mn);
        l_run[r] *= alpha;
        m_run[r] = mn;
#pragma unroll
        for (int c = 0; c < 4; ++c) o[c][r] *= alpha;
      }
      // relayout P (C layout -> A fragment) through per-wave LDS
      {
        const int col = lane & 15;
        const int rhalf = (lane >> 4) * 8;
#pragma unroll
        for (int r = 0; r < 8; ++r) {
          const int rowl = r + rhalf;
          pw[rowl * 32 + col] = f2bf(p0[r]);
          pw[rowl * 32 + 16 + col] = f2bf(p1[r]);
        }
      }
      __builtin_amdgcn_wave_barrier();  // keep ds stores before ds loads (in-order DS pipe)
      v16bf pf = ld_frag(pw + (lane & 15) * 32 + (lane >> 4) * 8);
      // row sums of P via WMMA: every lane of each 16-lane half gets its row's
      // sum in C layout -> directly matches l_run[r]'s broadcast layout.
      v8f rsum = wmma_bf16(pf, onesf, vzero());
#pragma unroll
      for (int r = 0; r < 8; ++r) l_run[r] += rsum[r];
#pragma unroll
      for (int c = 0; c < 4; ++c) {
        v16bf vf = ld_frag(Vt + (c * 16 + (lane & 15)) * VSTR + t0 + (lane >> 4) * 8);
        o[c] = wmma_bf16(pf, vf, o[c]);
      }
    }
    // write normalized output into [B,N,D] bf16 (heads merged)
#pragma unroll
    for (int r = 0; r < 8; ++r) {
      const float inv = 1.0f / l_run[r];
      const int row = rowbase + r + 8 * (lane >> 4);
      const int n = m * CH + row;
#pragma unroll
      for (int c = 0; c < 4; ++c) {
        const int col = h * DKH + c * 16 + (lane & 15);
        out_bf[((size_t)b * NSEQ + n) * DDIM + col] = f2bf(o[c][r] * inv);
      }
    }
  }
}

// ---------- host launch ----------
extern "C" void kernel_launch(void* const* d_in, const int* in_sizes, int n_in,
                              void* d_out, int out_size, void* d_ws, size_t ws_size,
                              hipStream_t stream) {
  const float* x  = (const float*)d_in[0];
  const float* Wq = (const float*)d_in[1];
  const float* bq = (const float*)d_in[2];
  const float* Wk = (const float*)d_in[3];
  const float* bk = (const float*)d_in[4];
  const float* Wv = (const float*)d_in[5];
  const float* bv = (const float*)d_in[6];
  const float* Wo = (const float*)d_in[7];
  const float* bo = (const float*)d_in[8];

  // ws layout (bytes): Wt 8MiB | xh 32MiB | QKV bf16 96MiB | attn_out bf16 32MiB
  const size_t QKV_ELEMS = (size_t)BSZ * HEADS * NCH * CH * DKH;  // 16,777,216
  unsigned short* wt   = (unsigned short*)d_ws;
  unsigned short* xh   = (unsigned short*)((char*)d_ws + (8ull << 20));
  unsigned short* qkv  = (unsigned short*)((char*)d_ws + (40ull << 20));
  unsigned short* aout = (unsigned short*)((char*)d_ws + (40ull << 20) + 3ull * QKV_ELEMS * 2);

  conv_x<<<dim3(8192), 256, 0, stream>>>(x, xh);
  prep_weights<<<dim3(4096, 4, 1), 256, 0, stream>>>(Wq, Wk, Wv, Wo, wt);

  gemm_kernel<0><<<dim3(8, 128, 3), 256, 0, stream>>>(
      xh, wt, bq, bk, bv, qkv, nullptr);

  attn_kernel<<<dim3(BSZ * HEADS * NCH), 256, 0, stream>>>(
      qkv, qkv + QKV_ELEMS, qkv + 2 * QKV_ELEMS, aout);

  gemm_kernel<1><<<dim3(8, 128, 1), 256, 0, stream>>>(
      aout, wt, bo, nullptr, nullptr, nullptr, (float*)d_out);
}
